// OdeRNN_64974265254323
// MI455X (gfx1250) — matrix operand order repaired and checked by reading, not compiled, not run on hardware
//
#include <hip/hip_runtime.h>
#include <hip/hip_bf16.h>

typedef __attribute__((ext_vector_type(16))) _Float16 v16h;
typedef __attribute__((ext_vector_type(8)))  _Float16 v8h;
typedef __attribute__((ext_vector_type(8)))  float    v8f;

#define T_STEPS   100
#define BATCH     8192
#define DIN       32
#define LDIM      64
#define HDIM      128
#define NSUB      4

#define WAVES_PER_BLOCK 8
#define ROWS_PER_WAVE   16
#define BLOCK_THREADS   (WAVES_PER_BLOCK * 32)
#define ROWS_PER_BLOCK  (WAVES_PER_BLOCK * ROWS_PER_WAVE)

// ---------------- LDS layout (bytes) ----------------
// Weight B-operand fragments: each fragment = 32 lanes * 32B = 1024B,
// holding a 32(K) x 16(N) f16 tile in this file's contiguous-K B order
// (lane<16: col=lane, K=0..15 ; lane>=16: col=lane-16, K=16..31).
#define FRAG_BYTES 1024
#define OFF_W1    0                          // 2 ktiles * 8 ntiles  (64x128)
#define OFF_W2    (OFF_W1  + 16 * FRAG_BYTES) // 4 ktiles * 4 ntiles  (128x64)
#define OFF_WIH   (OFF_W2  + 16 * FRAG_BYTES) // 1 ktile  * 12 ntiles (32x192)
#define OFF_WHH   (OFF_WIH + 12 * FRAG_BYTES) // 2 ktiles * 12 ntiles (64x192)
#define OFF_STAGE (OFF_WHH + 24 * FRAG_BYTES)
// Per-wave activation staging: 16 rows, stride 272B (256B data + 16B pad to
// spread rows across banks: row m starts at bank 4m mod 64). Row-major f16.
#define STAGE_STRIDE 272
#define STAGE_BYTES  (16 * STAGE_STRIDE)
#define X_COL_OFF    128   // byte offset of the x staging region within a row
#define LDS_TOTAL (OFF_STAGE + WAVES_PER_BLOCK * STAGE_BYTES)

// ---------------- small helpers ----------------
__device__ __forceinline__ void lds_fence() {
  asm volatile("s_wait_dscnt 0" ::: "memory");
}

__device__ __forceinline__ float sigf(float x) {
  return 1.0f / (1.0f + __expf(-x));
}

// D = A*B + C, f16 inputs, f32 accumulate (v_wmma_f32_16x16x32_f16)
__device__ __forceinline__ v8f wmma(v16h a, v16h b, v8f c) {
  return __builtin_amdgcn_wmma_f32_16x16x32_f16(
      /*neg_a=*/false, a, /*neg_b=*/false, b,
      /*c_mod=*/(short)0, c, /*reuse_a=*/false, /*reuse_b=*/false);
}

// Load one pre-swizzled B fragment (32B contiguous per lane).
__device__ __forceinline__ v16h ldb(const char* base, int frag, int lane) {
  return *(const v16h*)(base + frag * FRAG_BYTES + lane * 32);
}

// Load a 16x32 f16 A fragment from a row-major f16 staging region.
// A layout (matches B fragments): lanes 0-15 (M=0..15) hold K=0..15;
// lanes 16-31 hold K=16..31.
__device__ __forceinline__ v16h load_afrag(const char* stage, int lane, int colbyte) {
  const int row = lane & 15;
  const int hb  = (lane >> 4) << 5;  // +32B for upper half-lanes
  const char* p = stage + row * STAGE_STRIDE + colbyte + hb;
  v8h lo = *(const v8h*)(p);
  v8h hi = *(const v8h*)(p + 16);
  return __builtin_shufflevector(lo, hi, 0,1,2,3,4,5,6,7,8,9,10,11,12,13,14,15);
}

// Store a 16x16 f32 C tile (C layout: lane<16: N=lane,M=r ; lane>=16: N=lane-16,M=8+r)
// into the row-major f16 staging region at byte column `colbyte`.
__device__ __forceinline__ void store_ctile_f16(char* stage, int lane, int colbyte, v8f c) {
  const int n  = lane & 15;
  const int mb = (lane >> 4) << 3;
#pragma unroll
  for (int r = 0; r < 8; ++r) {
    *(_Float16*)(stage + (mb + r) * STAGE_STRIDE + colbyte + n * 2) = (_Float16)c[r];
  }
}

// One-time cooperative fill of B fragments from a row-major [N][K] fp32 weight
// (used as X @ W^T, i.e. B[k][n] = W[n][k]), converted to f16.
__device__ void fill_frags(char* dst, const float* __restrict__ src,
                           int K, int Kt, int Nt, int tid) {
  const int total = Kt * Nt * 512;  // f16 elements
  for (int i = tid; i < total; i += BLOCK_THREADS) {
    const int v  = i & 15;
    const int s  = (i >> 4) & 31;
    const int f  = i >> 9;
    const int nt = f % Nt;
    const int kt = f / Nt;
    const int n  = nt * 16 + (s & 15);
    const int k  = kt * 32 + ((s >> 4) << 4) + v;
    ((_Float16*)dst)[i] = (_Float16)src[n * K + k];
  }
}

// ---------------- ode_f: k = tanh(y @ W1^T + b1) @ W2^T + b2 ----------------
__device__ __forceinline__ void ode_f(const v8f y[4], v8f k[4], char* stage,
                                      const char* w1f, const char* w2f,
                                      const float* b1v, const float* b2v, int lane) {
  // stage y (16x64) as f16 row-major, then pull A fragments
#pragma unroll
  for (int j = 0; j < 4; ++j) store_ctile_f16(stage, lane, 32 * j, y[j]);
  lds_fence();
  v16h Ay0 = load_afrag(stage, lane, 0);
  v16h Ay1 = load_afrag(stage, lane, 64);
  lds_fence();
  // GEMM1 (M16 N128 K64) + tanh, staging u tiles as they are produced
#pragma unroll
  for (int nt = 0; nt < 8; ++nt) {
    v8f c;
#pragma unroll
    for (int i = 0; i < 8; ++i) c[i] = b1v[nt];
    c = wmma(Ay0, ldb(w1f, 0 * 8 + nt, lane), c);
    c = wmma(Ay1, ldb(w1f, 1 * 8 + nt, lane), c);
#pragma unroll
    for (int i = 0; i < 8; ++i) c[i] = tanhf(c[i]);
    store_ctile_f16(stage, lane, 32 * nt, c);
  }
  lds_fence();
  v16h Au[4];
#pragma unroll
  for (int kt = 0; kt < 4; ++kt) Au[kt] = load_afrag(stage, lane, 64 * kt);
  // GEMM2 (M16 N64 K128)
#pragma unroll
  for (int nt = 0; nt < 4; ++nt) {
    v8f c;
#pragma unroll
    for (int i = 0; i < 8; ++i) c[i] = b2v[nt];
#pragma unroll
    for (int kt = 0; kt < 4; ++kt) c = wmma(Au[kt], ldb(w2f, kt * 4 + nt, lane), c);
    k[nt] = c;
  }
}

// ---------------- RK4 with NSUB substeps, all in fp32 state ----------------
__device__ __forceinline__ void rk4(v8f h[4], float t0, float t1, char* stage,
                                    const char* w1f, const char* w2f,
                                    const float* b1v, const float* b2v, int lane) {
  const float dt  = (t1 - t0) * (1.0f / NSUB);
  const float dt6 = dt * (1.0f / 6.0f);
  const float dt3 = dt * (1.0f / 3.0f);
  const float dth = dt * 0.5f;
#pragma unroll 1
  for (int s = 0; s < NSUB; ++s) {
    v8f k[4], y[4], acc[4];
    ode_f(h, k, stage, w1f, w2f, b1v, b2v, lane);  // k1
#pragma unroll
    for (int j = 0; j < 4; ++j)
#pragma unroll
      for (int i = 0; i < 8; ++i) {
        acc[j][i] = h[j][i] + dt6 * k[j][i];
        y[j][i]   = h[j][i] + dth * k[j][i];
      }
    ode_f(y, k, stage, w1f, w2f, b1v, b2v, lane);  // k2
#pragma unroll
    for (int j = 0; j < 4; ++j)
#pragma unroll
      for (int i = 0; i < 8; ++i) {
        acc[j][i] += dt3 * k[j][i];
        y[j][i]    = h[j][i] + dth * k[j][i];
      }
    ode_f(y, k, stage, w1f, w2f, b1v, b2v, lane);  // k3
#pragma unroll
    for (int j = 0; j < 4; ++j)
#pragma unroll
      for (int i = 0; i < 8; ++i) {
        acc[j][i] += dt3 * k[j][i];
        y[j][i]    = h[j][i] + dt * k[j][i];
      }
    ode_f(y, k, stage, w1f, w2f, b1v, b2v, lane);  // k4
#pragma unroll
    for (int j = 0; j < 4; ++j)
#pragma unroll
      for (int i = 0; i < 8; ++i) h[j][i] = acc[j][i] + dt6 * k[j][i];
  }
}

// ---------------- x_t load: NaN mask + impute + stage + A fragment ----------
__device__ __forceinline__ v16h load_x(const float* __restrict__ xg, int row0,
                                       int lane, char* stage, unsigned* obsmask) {
  const int r    = lane & 15;
  const int half = lane >> 4;
  const float4* p = (const float4*)(xg + (size_t)(row0 + r) * DIN + half * 16);
  float4 q0 = p[0], q1 = p[1], q2 = p[2], q3 = p[3];
  float v[16] = {q0.x, q0.y, q0.z, q0.w, q1.x, q1.y, q1.z, q1.w,
                 q2.x, q2.y, q2.z, q2.w, q3.x, q3.y, q3.z, q3.w};
  bool has_nan = false;
  v8h lo, hi;
#pragma unroll
  for (int i = 0; i < 16; ++i) {
    float x = v[i];
    bool nn = (x != x);
    has_nan = has_nan || nn;
    x = nn ? 0.0f : x;
    if (i < 8) lo[i] = (_Float16)x; else hi[i - 8] = (_Float16)x;
  }
  unsigned bal = (unsigned)__ballot(has_nan);  // wave32: bits [31:0]
  *obsmask = (~(bal | (bal >> 16))) & 0xFFFFu; // bit m = row m fully observed
  char* sp = stage + r * STAGE_STRIDE + X_COL_OFF + half * 32;
  *(v8h*)sp        = lo;
  *(v8h*)(sp + 16) = hi;
  lds_fence();
  return load_afrag(stage, lane, X_COL_OFF);
}

// ---------------- GRU update with per-row mask select ----------------------
__device__ __forceinline__ void gru_update(v8f h[4], v16h Ax, unsigned obsmask,
                                           char* stage, const char* wihf,
                                           const char* whhf, const float* brz,
                                           const float* bin4, const float* bhn4,
                                           int lane) {
  // stage h_ode and pull its A fragments (x fragment already in Ax)
#pragma unroll
  for (int j = 0; j < 4; ++j) store_ctile_f16(stage, lane, 32 * j, h[j]);
  lds_fence();
  v16h Ah0 = load_afrag(stage, lane, 0);
  v16h Ah1 = load_afrag(stage, lane, 64);
  const int rowbase = (lane >> 4) << 3;
#pragma unroll
  for (int j = 0; j < 4; ++j) {
    // r / z gates: input+hidden contributions share one accumulator
    v8f gr;
#pragma unroll
    for (int i = 0; i < 8; ++i) gr[i] = brz[j];
    gr = wmma(Ah0, ldb(whhf, 0 * 12 + j, lane), gr);
    gr = wmma(Ah1, ldb(whhf, 1 * 12 + j, lane), gr);
    gr = wmma(Ax,  ldb(wihf, j, lane), gr);
    v8f gz;
#pragma unroll
    for (int i = 0; i < 8; ++i) gz[i] = brz[4 + j];
    gz = wmma(Ah0, ldb(whhf, 0 * 12 + 4 + j, lane), gz);
    gz = wmma(Ah1, ldb(whhf, 1 * 12 + 4 + j, lane), gz);
    gz = wmma(Ax,  ldb(wihf, 4 + j, lane), gz);
    // n gate needs i_n and h_n separately (r multiplies only h_n)
    v8f gin;
#pragma unroll
    for (int i = 0; i < 8; ++i) gin[i] = bin4[j];
    gin = wmma(Ax, ldb(wihf, 8 + j, lane), gin);
    v8f ghn;
#pragma unroll
    for (int i = 0; i < 8; ++i) ghn[i] = bhn4[j];
    ghn = wmma(Ah0, ldb(whhf, 0 * 12 + 8 + j, lane), ghn);
    ghn = wmma(Ah1, ldb(whhf, 1 * 12 + 8 + j, lane), ghn);
#pragma unroll
    for (int i = 0; i < 8; ++i) {
      float rg = sigf(gr[i]);
      float zg = sigf(gz[i]);
      float ng = tanhf(gin[i] + rg * ghn[i]);
      float hn = (1.0f - zg) * ng + zg * h[j][i];
      bool obs = (obsmask >> (rowbase + i)) & 1u;
      h[j][i] = obs ? hn : h[j][i];
    }
  }
}

// ---------------- persistent kernel ----------------------------------------
__global__ void __launch_bounds__(BLOCK_THREADS)
odernn_kernel(const float* __restrict__ ts, const float* __restrict__ obs_x,
              const float* __restrict__ W_ih, const float* __restrict__ W_hh,
              const float* __restrict__ b_ih, const float* __restrict__ b_hh,
              const float* __restrict__ W1, const float* __restrict__ b1,
              const float* __restrict__ W2, const float* __restrict__ b2,
              float* __restrict__ out) {
  extern __shared__ char smem[];
  const int tid  = threadIdx.x;
  const int lane = tid & 31;
  const int wave = tid >> 5;

  // one-time: swizzle all weights into LDS B-fragment layout (f16)
  fill_frags(smem + OFF_W1,  W1,   LDIM, 2, 8,  tid);
  fill_frags(smem + OFF_W2,  W2,   HDIM, 4, 4,  tid);
  fill_frags(smem + OFF_WIH, W_ih, DIN,  1, 12, tid);
  fill_frags(smem + OFF_WHH, W_hh, LDIM, 2, 12, tid);
  __syncthreads();

  const char* w1f  = smem + OFF_W1;
  const char* w2f  = smem + OFF_W2;
  const char* wihf = smem + OFF_WIH;
  const char* whhf = smem + OFF_WHH;
  char* stage = smem + OFF_STAGE + wave * STAGE_BYTES;

  // per-lane bias values (column = lane & 15 of each 16-wide N tile)
  const int c = lane & 15;
  float b1v[8], b2v[4], brz[8], bin4[4], bhn4[4];
#pragma unroll
  for (int nt = 0; nt < 8; ++nt) b1v[nt] = b1[nt * 16 + c];
#pragma unroll
  for (int nt = 0; nt < 4; ++nt) b2v[nt] = b2[nt * 16 + c];
#pragma unroll
  for (int nt = 0; nt < 8; ++nt) brz[nt] = b_ih[nt * 16 + c] + b_hh[nt * 16 + c];
#pragma unroll
  for (int j = 0; j < 4; ++j) {
    bin4[j] = b_ih[128 + j * 16 + c];
    bhn4[j] = b_hh[128 + j * 16 + c];
  }

  const int row0 = blockIdx.x * ROWS_PER_BLOCK + wave * ROWS_PER_WAVE;

  // hidden state: 4 x (16x16 f32 C tiles) = 16 rows x 64 cols, lives in VGPRs
  v8f h[4];
#pragma unroll
  for (int j = 0; j < 4; ++j)
#pragma unroll
    for (int i = 0; i < 8; ++i) h[j][i] = 0.0f;

  float tprev = ts[0];
#pragma unroll 1
  for (int t = 0; t < T_STEPS; ++t) {
    if (t > 0) {
      float tcur = ts[t];
      rk4(h, tprev, tcur, stage, w1f, w2f, b1v, b2v, lane);
      tprev = tcur;
    }
    if (t + 1 < T_STEPS) {  // global_prefetch_b8 for next timestep's slab
      __builtin_prefetch(
          obs_x + ((size_t)(t + 1) * BATCH + row0 + (lane & 15)) * DIN + (lane >> 4) * 16,
          0, 3);
    }
    unsigned obsmask;
    v16h Ax = load_x(obs_x + (size_t)t * BATCH * DIN, row0, lane, stage, &obsmask);
    gru_update(h, Ax, obsmask, stage, wihf, whhf, brz, bin4, bhn4, lane);
  }

  // write h_final (B x L fp32) from C layout
#pragma unroll
  for (int j = 0; j < 4; ++j) {
    const int n = 16 * j + (lane & 15);
#pragma unroll
    for (int r = 0; r < 8; ++r) {
      const int m = ((lane >> 4) << 3) + r;
      out[(size_t)(row0 + m) * LDIM + n] = h[j][r];
    }
  }
}

extern "C" void kernel_launch(void* const* d_in, const int* in_sizes, int n_in,
                              void* d_out, int out_size, void* d_ws, size_t ws_size,
                              hipStream_t stream) {
  (void)in_sizes; (void)n_in; (void)out_size; (void)d_ws; (void)ws_size;
  const float* ts   = (const float*)d_in[0];
  const float* obsx = (const float*)d_in[1];
  const float* W_ih = (const float*)d_in[2];
  const float* W_hh = (const float*)d_in[3];
  const float* b_ih = (const float*)d_in[4];
  const float* b_hh = (const float*)d_in[5];
  const float* W1   = (const float*)d_in[6];
  const float* b1   = (const float*)d_in[7];
  const float* W2   = (const float*)d_in[8];
  const float* b2   = (const float*)d_in[9];
  float* out = (float*)d_out;

  dim3 grid(BATCH / ROWS_PER_BLOCK);   // 64 blocks x 8 independent waves
  dim3 block(BLOCK_THREADS);
  odernn_kernel<<<grid, block, LDS_TOTAL, stream>>>(
      ts, obsx, W_ih, W_hh, b_ih, b_hh, W1, b1, W2, b2, out);
}
